// NerModel_42314017800267
// MI455X (gfx1250) — compile-verified, matrix operand order checked by reading, not run on hardware
//
#include <hip/hip_runtime.h>
#include <hip/hip_bf16.h>

typedef __attribute__((ext_vector_type(16))) _Float16 v16h;
typedef __attribute__((ext_vector_type(8)))  _Float16 v8h;
typedef __attribute__((ext_vector_type(8)))  float    v8f;

#define B_  64
#define S_  512
#define E_  300
#define EP  320          // E padded to multiple of 32 (WMMA K)
#define H_  256
#define G4  1024         // 4*H
#define C_  32
#define H2  512          // 2*H

__device__ __forceinline__ float sigf(float x) { return 1.0f / (1.0f + __expf(-x)); }
// fast tanh on v_exp_f32: saturates correctly at +/-1 for large |x|
__device__ __forceinline__ float ftanh(float x) {
    float e = __expf(2.0f * x);
    return 1.0f - 2.0f / (e + 1.0f);
}

__device__ __forceinline__ v16h cat8(v8h a, v8h b) {
    return __builtin_shufflevector(a, b, 0,1,2,3,4,5,6,7,8,9,10,11,12,13,14,15);
}

// ---------------------------------------------------------------------------
// Gather embeddings -> f16, K padded to EP with zeros. x16[b*S+s][0..EP)
// ---------------------------------------------------------------------------
__global__ void k_gather(const int* __restrict__ tok, const float* __restrict__ emb,
                         _Float16* __restrict__ x16) {
    int idx = blockIdx.x * blockDim.x + threadIdx.x;
    if (idx >= B_ * S_ * EP) return;
    int e = idx % EP;
    int bs = idx / EP;
    float v = 0.0f;
    if (e < E_) v = emb[(long long)tok[bs] * E_ + e];
    x16[idx] = (_Float16)v;
}

// ---------------------------------------------------------------------------
// Pack a row-major (K x N) f32 weight into B-fragment layout:
//   out[((ntile*Kt + ktile)*32 + lane)*16 + e]
//   lane<16:  n = nt*16+lane,     k = kt*32 +   e + (e>=8 ? 8:0)
//   lane>=16: n = nt*16+lane-16,  k = kt*32 + 8+e + (e>=8 ? 8:0)
// K rows beyond Kreal are zero-padded.
// ---------------------------------------------------------------------------
__global__ void k_pack(const float* __restrict__ W, _Float16* __restrict__ out,
                       int Kreal, int Kt, int Nt, int N) {
    int idx = blockIdx.x * blockDim.x + threadIdx.x;
    int total = Kt * Nt * 512;
    if (idx >= total) return;
    int e    = idx & 15;
    int lane = (idx >> 4) & 31;
    int rest = idx >> 9;
    int kt = rest % Kt;
    int nt = rest / Kt;
    int k = kt * 32 + ((lane >= 16) ? 8 : 0) + e + ((e >= 8) ? 8 : 0);
    int n = nt * 16 + (lane & 15);
    float v = (k < Kreal) ? W[(long long)k * N + n] : 0.0f;
    out[idx] = (_Float16)v;
}

// ---------------------------------------------------------------------------
// xproj = x @ W + bias   (M = B*S = 32768, N = 1024, Kp = 320)
// One wave per (m-tile, n-quad): 16x64 output, 10 K-steps, 40 WMMAs.
// rev selects reversed time order of x rows (backward LSTM input).
// ---------------------------------------------------------------------------
__global__ __launch_bounds__(256)
void k_xproj(const _Float16* __restrict__ x16, const _Float16* __restrict__ Wp,
             const float* __restrict__ bias, float* __restrict__ out, int rev) {
    int wid    = (blockIdx.x * blockDim.x + threadIdx.x) >> 5;
    int lane   = threadIdx.x & 31;
    int lanelo = lane & 15;
    int hi     = lane >> 4;
    int mt = wid >> 4;        // 0..2047
    int nq = wid & 15;        // 0..15 : 64 columns
    int m_base = mt * 16;

    long long arow;
    {
        int m = m_base + lanelo;
        int b = m / S_, s = m % S_;
        arow = rev ? ((long long)b * S_ + (S_ - 1 - s)) : (long long)m;
    }
    const _Float16* abase = x16 + arow * EP;

    const int Kt = EP / 32;   // 10
    v8f acc[4] = {};
    for (int k = 0; k < Kt; ++k) {
        int kb = k * 32 + hi * 8;
        v8h a0 = *(const v8h*)(abase + kb);
        v8h a1 = *(const v8h*)(abase + kb + 16);
        v16h A = cat8(a0, a1);
        if (k + 1 < Kt) __builtin_prefetch(abase + kb + 32, 0, 1);
#pragma unroll
        for (int nt = 0; nt < 4; ++nt) {
            int ntile = nq * 4 + nt;
            const _Float16* pb = Wp + ((long long)(ntile * Kt + k) * 32 + lane) * 16;
            v8h b0 = *(const v8h*)pb;
            v8h b1 = *(const v8h*)(pb + 8);
            v16h Bf = cat8(b0, b1);
            acc[nt] = __builtin_amdgcn_wmma_f32_16x16x32_f16(
                false, A, false, Bf, (short)0, acc[nt], false, false);
        }
    }
#pragma unroll
    for (int nt = 0; nt < 4; ++nt) {
        int n = nq * 64 + nt * 16 + lanelo;
        float bv = bias[n];
#pragma unroll
        for (int r = 0; r < 8; ++r) {
            int m = m_base + r + hi * 8;
            out[(long long)m * G4 + n] = acc[nt][r] + bv;
        }
    }
}

// ---------------------------------------------------------------------------
// Persistent LSTM recurrence. 8 blocks = 2 dirs x 4 batch-chunks of 16.
// blockDim = 512 (16 waves); wave w owns hidden cols [w*16, w*16+16).
// Per step: acc <- xproj[t] (software-pipelined one step ahead so the fetch
// latency hides behind the previous step's 32 WMMAs + gate math + barrier),
// 8 K-steps x 4 gates WMMA against h_{t-1} (f16 in LDS, double-buffered),
// fused gate math in regs, c kept in registers in D-fragment layout.
// ---------------------------------------------------------------------------
__global__ __launch_bounds__(512)
void k_lstm(const _Float16* __restrict__ Ufp, const _Float16* __restrict__ Ubp,
            const float* __restrict__ xpf, const float* __restrict__ xpb,
            float* __restrict__ hf, float* __restrict__ hb) {
    int dir   = blockIdx.x >> 2;
    int chunk = blockIdx.x & 3;
    const _Float16* Up = dir ? Ubp : Ufp;
    const float*    xp = dir ? xpb : xpf;
    float*        hout = dir ? hb  : hf;

    int wid    = threadIdx.x >> 5;    // 0..15
    int lane   = threadIdx.x & 31;
    int lanelo = lane & 15;
    int hi     = lane >> 4;
    int j      = wid * 16 + lanelo;   // hidden column

    __shared__ __align__(16) _Float16 hbuf[2][16][264]; // pad row to kill bank conflicts

    for (int i = threadIdx.x; i < 16 * 264; i += blockDim.x)
        hbuf[0][i / 264][i % 264] = (_Float16)0.0f;
    __syncthreads();

    float c[8];
#pragma unroll
    for (int r = 0; r < 8; ++r) c[r] = 0.0f;

    // prologue: fetch xproj for t=0
    v8f xnext[4];
#pragma unroll
    for (int g = 0; g < 4; ++g) {
        int n = g * H_ + j;
#pragma unroll
        for (int r = 0; r < 8; ++r) {
            int bg = chunk * 16 + r + hi * 8;
            xnext[g][r] = xp[((long long)bg * S_ + 0) * G4 + n];
        }
    }

    const int Kt = H_ / 32;           // 8
    for (int t = 0; t < S_; ++t) {
        int pb = t & 1, nb = pb ^ 1;

        // consume prefetched xproj[t]; immediately issue loads for t+1 so the
        // s_wait lands after this step's WMMAs + gates + barrier
        v8f acc[4];
#pragma unroll
        for (int g = 0; g < 4; ++g) acc[g] = xnext[g];

        int t1 = (t + 1 < S_) ? (t + 1) : t;
#pragma unroll
        for (int g = 0; g < 4; ++g) {
            int n = g * H_ + j;
#pragma unroll
            for (int r = 0; r < 8; ++r) {
                int bg = chunk * 16 + r + hi * 8;
                xnext[g][r] = xp[((long long)bg * S_ + t1) * G4 + n];
            }
        }

#pragma unroll
        for (int k = 0; k < Kt; ++k) {
            int kb = k * 32 + hi * 8;
            v8h a0 = *(const v8h*)&hbuf[pb][lanelo][kb];
            v8h a1 = *(const v8h*)&hbuf[pb][lanelo][kb + 16];
            v16h A = cat8(a0, a1);
#pragma unroll
            for (int g = 0; g < 4; ++g) {
                int ntile = g * 16 + wid;
                const _Float16* pw = Up + ((long long)(ntile * Kt + k) * 32 + lane) * 16;
                v8h b0 = *(const v8h*)pw;
                v8h b1 = *(const v8h*)(pw + 8);
                v16h Bf = cat8(b0, b1);
                acc[g] = __builtin_amdgcn_wmma_f32_16x16x32_f16(
                    false, A, false, Bf, (short)0, acc[g], false, false);
            }
        }
        int s_out = dir ? (S_ - 1 - t) : t;
#pragma unroll
        for (int r = 0; r < 8; ++r) {
            float iv = sigf(acc[0][r]);
            float fv = sigf(acc[1][r]);
            float gv = ftanh(acc[2][r]);
            float ov = sigf(acc[3][r]);
            float cn = fv * c[r] + iv * gv;
            c[r] = cn;
            float hv = ov * ftanh(cn);
            int mrow = r + hi * 8;
            int bg = chunk * 16 + mrow;
            hout[((long long)bg * S_ + s_out) * H_ + j] = hv;
            hbuf[nb][mrow][j] = (_Float16)hv;
        }
        __syncthreads();
    }
}

// ---------------------------------------------------------------------------
// logits = concat(hf,hb) @ Wd + bd  (M=32768, N=32, K=512). One wave per
// m-tile, 16 K-steps, 2 n-tiles -> 32 WMMAs/wave. A built on-the-fly f32->f16.
// ---------------------------------------------------------------------------
__global__ __launch_bounds__(256)
void k_logits(const float* __restrict__ hf, const float* __restrict__ hb,
              const _Float16* __restrict__ Wdp, const float* __restrict__ bd,
              float* __restrict__ out) {
    int wid    = (blockIdx.x * blockDim.x + threadIdx.x) >> 5;  // m-tile 0..2047
    int lane   = threadIdx.x & 31;
    int lanelo = lane & 15;
    int hi     = lane >> 4;
    int row = wid * 16 + lanelo;                 // = b*S + s
    const float* hrf = hf + (long long)row * H_;
    const float* hrb = hb + (long long)row * H_;

    const int Kt = H2 / 32;                      // 16
    v8f acc[2] = {};
    for (int kk = 0; kk < Kt; ++kk) {
        int col = kk * 32 + hi * 8;              // both 8-chunks stay in same half
        const float* src = (col < H_) ? (hrf + col) : (hrb + (col - H_));
        v16h A;
#pragma unroll
        for (int e = 0; e < 8; ++e) {
            A[e]     = (_Float16)src[e];
            A[8 + e] = (_Float16)src[16 + e];
        }
#pragma unroll
        for (int nt = 0; nt < 2; ++nt) {
            const _Float16* pw = Wdp + ((long long)(nt * Kt + kk) * 32 + lane) * 16;
            v8h b0 = *(const v8h*)pw;
            v8h b1 = *(const v8h*)(pw + 8);
            v16h Bf = cat8(b0, b1);
            acc[nt] = __builtin_amdgcn_wmma_f32_16x16x32_f16(
                false, A, false, Bf, (short)0, acc[nt], false, false);
        }
    }
#pragma unroll
    for (int nt = 0; nt < 2; ++nt) {
        int n = nt * 16 + lanelo;
        float bv = bd[n];
#pragma unroll
        for (int r = 0; r < 8; ++r) {
            int m = wid * 16 + r + hi * 8;
            out[(long long)m * C_ + n] = acc[nt][r] + bv;
        }
    }
}

// ---------------------------------------------------------------------------
// CRF log-likelihood: one wave32 per batch row, lane = class. Alpha kept in a
// register; cross-lane logsumexp via wave shuffles (ds_permute).
// ---------------------------------------------------------------------------
__global__ __launch_bounds__(32)
void k_crf(const float* __restrict__ logits, const int* __restrict__ targets,
           const int* __restrict__ lengths, const float* __restrict__ trans,
           float* __restrict__ ll) {
    int b = blockIdx.x;
    int j = threadIdx.x;                     // class 0..31
    const float* lg = logits + (long long)b * S_ * C_;
    const int*   tg = targets + b * S_;
    int len = lengths[b];

    float un = 0.0f, bi = 0.0f;
    for (int t = j; t < S_; t += 32) {
        if (t < len)            un += lg[t * C_ + tg[t]];
        if (t >= 1 && t < len)  bi += trans[tg[t - 1] * C_ + tg[t]];
    }
    for (int off = 16; off; off >>= 1) {
        un += __shfl_xor(un, off, 32);
        bi += __shfl_xor(bi, off, 32);
    }

    float alpha = lg[j];
    for (int t = 1; t < S_; ++t) {
        if (t >= len) break;                 // mask is monotone in t
        float m = -3.4e38f, s = 0.0f;
        for (int i = 0; i < 32; ++i) {
            float ai = __shfl(alpha, i, 32);
            float v = ai + trans[i * C_ + j];
            float nm = fmaxf(m, v);
            s = s * __expf(m - nm) + __expf(v - nm);
            m = nm;
        }
        alpha = m + __logf(s) + lg[t * C_ + j];
    }
    float m = alpha;
    for (int off = 16; off; off >>= 1) m = fmaxf(m, __shfl_xor(m, off, 32));
    float s = __expf(alpha - m);
    for (int off = 16; off; off >>= 1) s += __shfl_xor(s, off, 32);
    if (j == 0) ll[b] = un + bi - (m + __logf(s));
}

__global__ void k_copytrans(const float* __restrict__ trans, float* __restrict__ out) {
    int i = blockIdx.x * blockDim.x + threadIdx.x;
    if (i < C_ * C_) out[i] = trans[i];
}

// ---------------------------------------------------------------------------
extern "C" void kernel_launch(void* const* d_in, const int* in_sizes, int n_in,
                              void* d_out, int out_size, void* d_ws, size_t ws_size,
                              hipStream_t stream) {
    const int*   tokens  = (const int*)  d_in[0];
    const int*   lengths = (const int*)  d_in[1];
    const int*   targets = (const int*)  d_in[2];
    const float* emb     = (const float*)d_in[3];
    const float* Wf      = (const float*)d_in[4];
    const float* Uf      = (const float*)d_in[5];
    const float* bf      = (const float*)d_in[6];
    const float* Wb      = (const float*)d_in[7];
    const float* Ub      = (const float*)d_in[8];
    const float* bb      = (const float*)d_in[9];
    const float* Wd      = (const float*)d_in[10];
    const float* bd      = (const float*)d_in[11];
    const float* trans   = (const float*)d_in[12];

    // workspace carve-up (256B aligned)
    char* p = (char*)d_ws;
    auto take = [&](size_t bytes) -> char* {
        char* r = p;
        p += (bytes + 255) & ~size_t(255);
        return r;
    };
    _Float16* x16   = (_Float16*)take((size_t)B_ * S_ * EP * 2);
    _Float16* Wfp   = (_Float16*)take((size_t)(EP / 32) * 64 * 512 * 2);
    _Float16* Wbp   = (_Float16*)take((size_t)(EP / 32) * 64 * 512 * 2);
    _Float16* Ufp   = (_Float16*)take((size_t)(H_ / 32) * 64 * 512 * 2);
    _Float16* Ubp   = (_Float16*)take((size_t)(H_ / 32) * 64 * 512 * 2);
    _Float16* Wdp   = (_Float16*)take((size_t)(H2 / 32) * 2 * 512 * 2);
    float*    xpf   = (float*)take((size_t)B_ * S_ * G4 * 4);
    float*    xpb   = (float*)take((size_t)B_ * S_ * G4 * 4);
    float*    hfbuf = (float*)take((size_t)B_ * S_ * H_ * 4);
    float*    hbbuf = (float*)take((size_t)B_ * S_ * H_ * 4);

    float* logits_out = (float*)d_out;
    float* ll_out     = logits_out + (size_t)B_ * S_ * C_;
    float* trans_out  = ll_out + B_;

    // 1) gather + f16 convert
    {
        int total = B_ * S_ * EP;
        k_gather<<<(total + 255) / 256, 256, 0, stream>>>(tokens, emb, x16);
    }
    // 2) weight packs into B-fragment layout
    k_pack<<<(10 * 64 * 512 + 255) / 256, 256, 0, stream>>>(Wf, Wfp, E_, 10, 64, G4);
    k_pack<<<(10 * 64 * 512 + 255) / 256, 256, 0, stream>>>(Wb, Wbp, E_, 10, 64, G4);
    k_pack<<<( 8 * 64 * 512 + 255) / 256, 256, 0, stream>>>(Uf, Ufp, H_,  8, 64, G4);
    k_pack<<<( 8 * 64 * 512 + 255) / 256, 256, 0, stream>>>(Ub, Ubp, H_,  8, 64, G4);
    k_pack<<<(16 *  2 * 512 + 255) / 256, 256, 0, stream>>>(Wd, Wdp, H2, 16,  2, C_);
    // 3) input projections (WMMA GEMM), fwd + reversed
    k_xproj<<<4096, 256, 0, stream>>>(x16, Wfp, bf, xpf, 0);
    k_xproj<<<4096, 256, 0, stream>>>(x16, Wbp, bb, xpb, 1);
    // 4) persistent bidirectional LSTM recurrence
    k_lstm<<<8, 512, 0, stream>>>(Ufp, Ubp, xpf, xpb, hfbuf, hbbuf);
    // 5) decoder GEMM -> logits (straight into d_out)
    k_logits<<<256, 256, 0, stream>>>(hfbuf, hbbuf, Wdp, bd, logits_out);
    // 6) CRF log-likelihood
    k_crf<<<B_, 32, 0, stream>>>(logits_out, targets, lengths, trans, ll_out);
    // 7) pass trans through as third output
    k_copytrans<<<4, 256, 0, stream>>>(trans, trans_out);
}
